// MapBasedMultiHeadAttention_75935021793386
// MI455X (gfx1250) — matrix operand order repaired and checked by reading, not compiled, not run on hardware
//
#include <hip/hip_runtime.h>
#include <hip/hip_bf16.h>

// MapBasedMultiHeadAttention for MI455X (gfx1250), wave32 + WMMA bf16.
// B=2, L=2048, D=1024, H=16, DK=DV=64.

#define Bn 2
#define Ln 2048
#define Dn 1024
#define Hn 16
#define NEGV (-1e10f)

typedef __attribute__((ext_vector_type(16))) __bf16 v16bf;
typedef __attribute__((ext_vector_type(8)))  float  v8f;

__device__ __forceinline__ __bf16 to_bf16(float x) {
  union { float f; unsigned u; } in; in.f = x;
  unsigned r = in.u + 0x7FFFu + ((in.u >> 16) & 1u);   // round-to-nearest-even
  union { unsigned short s; __bf16 b; } o; o.s = (unsigned short)(r >> 16);
  return o.b;
}

// A fragment (16x32 bf16, MxK): lane l -> row l&15, K = (l>>4)*16 + {0..15}
__device__ __forceinline__ v16bf load_a_frag(const __bf16* tile, int ldk, int lane) {
  int r = lane & 15, g = lane >> 4;
  const __bf16* base = tile + r * ldk + g * 16;
  union { v16bf v; uint4 u[2]; } f;
  f.u[0] = *(const uint4*)(base);
  f.u[1] = *(const uint4*)(base + 8);
  return f.v;
}

// B fragment (32x16 bf16, KxN) from transposed LDS tile Bt[n][k]:
// lane l -> col l&15, K = (l>>4)*16 + {0..15}  (16 contiguous bf16)
__device__ __forceinline__ v16bf load_b_frag(const __bf16* tileT, int ldk, int lane) {
  int c = lane & 15, g = lane >> 4;
  const __bf16* base = tileT + c * ldk + g * 16;
  union { v16bf v; uint4 u[2]; } f;
  f.u[0] = ((const uint4*)base)[0];
  f.u[1] = ((const uint4*)base)[1];
  return f.v;
}

// ---------------- QKV projection: Y[b][h][l][d] = (X @ W), head-major out ----
__global__ __launch_bounds__(256) void k_proj(
    const float* __restrict__ Xq, const float* __restrict__ Xk, const float* __restrict__ Xv,
    const float* __restrict__ Wq, const float* __restrict__ Wk, const float* __restrict__ Wv,
    float* __restrict__ Yq, float* __restrict__ Yk, float* __restrict__ Yv) {
  const float* X; const float* W; float* Y;
  if (blockIdx.z == 0)      { X = Xq; W = Wq; Y = Yq; }
  else if (blockIdx.z == 1) { X = Xk; W = Wk; Y = Yk; }
  else                      { X = Xv; W = Wv; Y = Yv; }
  const int K = 1024, N = 1024;
  const int m0 = blockIdx.y * 64, n0 = blockIdx.x * 64;
  __shared__ __bf16 As[64 * 48];   // [row][k], padded stride
  __shared__ __bf16 Bs[64 * 48];   // [n][k]  (transposed)
  const int tid = threadIdx.x, lane = tid & 31, wid = tid >> 5;
  const int msub = wid & 3, nsub0 = (wid >> 2) * 2;
  v8f acc0 = {0,0,0,0,0,0,0,0}, acc1 = {0,0,0,0,0,0,0,0};
  for (int kc = 0; kc < K; kc += 32) {
    __syncthreads();
    if (kc + 32 < K) __builtin_prefetch(X + (size_t)(m0 + (tid & 63)) * K + kc + 32);
#pragma unroll
    for (int j = 0; j < 2; ++j) {          // A: 64 rows x 8 float4
      int i = tid + 256 * j;
      int row = i >> 3, c4 = i & 7;
      float4 f = *(const float4*)(X + (size_t)(m0 + row) * K + kc + c4 * 4);
      __bf16* d = As + row * 48 + c4 * 4;
      d[0] = to_bf16(f.x); d[1] = to_bf16(f.y); d[2] = to_bf16(f.z); d[3] = to_bf16(f.w);
    }
#pragma unroll
    for (int j = 0; j < 2; ++j) {          // B: 32 k-rows x 16 float4, store transposed
      int i = tid + 256 * j;
      int kr = i >> 4, c4 = i & 15;
      float4 f = *(const float4*)(W + (size_t)(kc + kr) * N + n0 + c4 * 4);
      Bs[(c4 * 4 + 0) * 48 + kr] = to_bf16(f.x);
      Bs[(c4 * 4 + 1) * 48 + kr] = to_bf16(f.y);
      Bs[(c4 * 4 + 2) * 48 + kr] = to_bf16(f.z);
      Bs[(c4 * 4 + 3) * 48 + kr] = to_bf16(f.w);
    }
    __syncthreads();
    v16bf a  = load_a_frag(As + msub * 16 * 48, 48, lane);
    v16bf b0 = load_b_frag(Bs + nsub0 * 16 * 48, 48, lane);
    v16bf b1 = load_b_frag(Bs + (nsub0 + 1) * 16 * 48, 48, lane);
    acc0 = __builtin_amdgcn_wmma_f32_16x16x32_bf16(false, a, false, b0, (short)0, acc0, false, false);
    acc1 = __builtin_amdgcn_wmma_f32_16x16x32_bf16(false, a, false, b1, (short)0, acc1, false, false);
  }
  const int col = lane & 15, g = lane >> 4;
#pragma unroll
  for (int j = 0; j < 8; ++j) {            // C/D layout: row = j + 8*(l>>4), col = l&15
    int m = m0 + msub * 16 + j + 8 * g;
    int b = m >> 11, l = m & (Ln - 1);
    int na = n0 + nsub0 * 16 + col;
    int ha = na >> 6, da = na & 63;
    Y[(((size_t)(b * Hn + ha)) * Ln + l) * 64 + da] = acc0[j];
    int nb = na + 16;
    int hb = nb >> 6, db = nb & 63;
    Y[(((size_t)(b * Hn + hb)) * Ln + l) * 64 + db] = acc1[j];
  }
}

// ---------------- logits scalars: sq = qh.ws[:64], sk = kh.ws[64:] ----------
__global__ __launch_bounds__(256) void k_logit(
    const float* __restrict__ qh, const float* __restrict__ kh,
    const float* __restrict__ ws, float* __restrict__ sq, float* __restrict__ sk) {
  int t = blockIdx.x * 256 + threadIdx.x;
  const int BHL = Bn * Hn * Ln;            // 65536
  const float* row; const float* w; float* out; int idx;
  if (t < BHL) { idx = t;        row = qh + (size_t)idx * 64; w = ws;      out = sq; }
  else         { idx = t - BHL;  row = kh + (size_t)idx * 64; w = ws + 64; out = sk; }
  float s = 0.f;
#pragma unroll 8
  for (int d = 0; d < 64; ++d) s += row[d] * w[d];
  out[idx] = s;
}

// ------------- masked softmax (rank-1 logits) + attn out + ctx = P @ V ------
__global__ __launch_bounds__(256) void k_attn(
    const float* __restrict__ sq, const float* __restrict__ sk,
    const float* __restrict__ vh, const unsigned char* __restrict__ mask,
    float* __restrict__ attn, float* __restrict__ ctx) {
  const int bh = blockIdx.y, b = bh >> 4, h = bh & 15;
  const int q0 = blockIdx.x * 32;
  const int tid = threadIdx.x, lane = tid & 31, wid = tid >> 5;

  __shared__ float  s_sk[Ln];
  __shared__ float  s_m[32];
  __shared__ float  s_iz[32];
  __shared__ __bf16 Ps[32 * 48];
  __shared__ __bf16 Vs[64 * 48];

  const float* skb = sk + (size_t)bh * Ln;
  for (int i = tid; i < Ln; i += 256) s_sk[i] = skb[i];
  __syncthreads();

  const unsigned char* mb = mask + (size_t)b * Ln * Ln;

  // pass 1+2: per-row masked max and sum-exp (one wave handles 4 rows)
  for (int r = 0; r < 4; ++r) {
    int rq = wid * 4 + r, q = q0 + rq;
    const unsigned char* mrow = mb + (size_t)q * Ln;
    float sqv = sq[(size_t)bh * Ln + q];
    float mx = NEGV;
    for (int k = lane; k < Ln; k += 32)
      if (!mrow[k]) mx = fmaxf(mx, sqv + s_sk[k]);
#pragma unroll
    for (int off = 16; off > 0; off >>= 1) mx = fmaxf(mx, __shfl_xor(mx, off, 32));
    float z = 0.f;
    for (int k = lane; k < Ln; k += 32)
      if (!mrow[k]) z += __expf(sqv + s_sk[k] - mx);
#pragma unroll
    for (int off = 16; off > 0; off >>= 1) z += __shfl_xor(z, off, 32);
    if (lane == 0) {
      if (mx < -1e9f) { s_m[rq] = 0.f; s_iz[rq] = 0.f; }       // fully-masked row
      else            { s_m[rq] = mx - sqv; s_iz[rq] = 1.f / fmaxf(z, 1e-30f); }
    }
  }
  __syncthreads();

  // pass 3: stream attn (non-temporal) + accumulate ctx via WMMA
  const int msub = wid & 1, nsub = wid >> 1;       // 2x4 subtiles of 32x64
  v8f acc = {0,0,0,0,0,0,0,0};
  float* attn_bh = attn + (size_t)bh * Ln * Ln;
  const float* vhb = vh + (size_t)bh * Ln * 64;
  for (int kc = 0; kc < Ln; kc += 32) {
    __syncthreads();
#pragma unroll
    for (int j = 0; j < 4; ++j) {                  // P: 32 rows x 32 k
      int i = tid + 256 * j;
      int r = i >> 5, kk = i & 31;
      int q = q0 + r, k = kc + kk;
      float p = 0.f;
      if (!mb[(size_t)q * Ln + k]) p = __expf(s_sk[k] - s_m[r]) * s_iz[r];
      __builtin_nontemporal_store(p, attn_bh + (size_t)q * Ln + k);
      Ps[r * 48 + kk] = to_bf16(p);
    }
#pragma unroll
    for (int j = 0; j < 8; ++j) {                  // V chunk transposed: Vs[n][k]
      int i = tid + 256 * j;
      int kk = i >> 6, n = i & 63;
      Vs[n * 48 + kk] = to_bf16(vhb[(size_t)(kc + kk) * 64 + n]);
    }
    __syncthreads();
    v16bf a   = load_a_frag(Ps + msub * 16 * 48, 48, lane);
    v16bf bfr = load_b_frag(Vs + nsub * 16 * 48, 48, lane);
    acc = __builtin_amdgcn_wmma_f32_16x16x32_bf16(false, a, false, bfr, (short)0, acc, false, false);
  }
  const int col = lane & 15, g = lane >> 4;
#pragma unroll
  for (int j = 0; j < 8; ++j) {
    int q = q0 + msub * 16 + j + 8 * g;
    int d = nsub * 16 + col;
    ctx[((size_t)(b * Ln + q)) * Dn + h * 64 + d] = acc[j];
  }
}

// ---------------- out-proj: tmp = relu(ctx@Wo + bo) + q ---------------------
__global__ __launch_bounds__(256) void k_ogemm(
    const float* __restrict__ A, const float* __restrict__ W,
    const float* __restrict__ bo, const float* __restrict__ qin,
    float* __restrict__ Y) {
  const int K = 1024, N = 1024;
  const int m0 = blockIdx.y * 64, n0 = blockIdx.x * 64;
  __shared__ __bf16 As[64 * 48];
  __shared__ __bf16 Bs[64 * 48];
  const int tid = threadIdx.x, lane = tid & 31, wid = tid >> 5;
  const int msub = wid & 3, nsub0 = (wid >> 2) * 2;
  v8f acc0 = {0,0,0,0,0,0,0,0}, acc1 = {0,0,0,0,0,0,0,0};
  for (int kc = 0; kc < K; kc += 32) {
    __syncthreads();
    if (kc + 32 < K) __builtin_prefetch(A + (size_t)(m0 + (tid & 63)) * K + kc + 32);
#pragma unroll
    for (int j = 0; j < 2; ++j) {
      int i = tid + 256 * j;
      int row = i >> 3, c4 = i & 7;
      float4 f = *(const float4*)(A + (size_t)(m0 + row) * K + kc + c4 * 4);
      __bf16* d = As + row * 48 + c4 * 4;
      d[0] = to_bf16(f.x); d[1] = to_bf16(f.y); d[2] = to_bf16(f.z); d[3] = to_bf16(f.w);
    }
#pragma unroll
    for (int j = 0; j < 2; ++j) {
      int i = tid + 256 * j;
      int kr = i >> 4, c4 = i & 15;
      float4 f = *(const float4*)(W + (size_t)(kc + kr) * N + n0 + c4 * 4);
      Bs[(c4 * 4 + 0) * 48 + kr] = to_bf16(f.x);
      Bs[(c4 * 4 + 1) * 48 + kr] = to_bf16(f.y);
      Bs[(c4 * 4 + 2) * 48 + kr] = to_bf16(f.z);
      Bs[(c4 * 4 + 3) * 48 + kr] = to_bf16(f.w);
    }
    __syncthreads();
    v16bf a  = load_a_frag(As + msub * 16 * 48, 48, lane);
    v16bf b0 = load_b_frag(Bs + nsub0 * 16 * 48, 48, lane);
    v16bf b1 = load_b_frag(Bs + (nsub0 + 1) * 16 * 48, 48, lane);
    acc0 = __builtin_amdgcn_wmma_f32_16x16x32_bf16(false, a, false, b0, (short)0, acc0, false, false);
    acc1 = __builtin_amdgcn_wmma_f32_16x16x32_bf16(false, a, false, b1, (short)0, acc1, false, false);
  }
  const int col = lane & 15, g = lane >> 4;
#pragma unroll
  for (int j = 0; j < 8; ++j) {
    int m = m0 + msub * 16 + j + 8 * g;
    int na = n0 + nsub0 * 16 + col;
    float xa = fmaxf(acc0[j] + bo[na], 0.f) + qin[(size_t)m * Dn + na];
    Y[(size_t)m * Dn + na] = xa;
    int nb = na + 16;
    float xb = fmaxf(acc1[j] + bo[nb], 0.f) + qin[(size_t)m * Dn + nb];
    Y[(size_t)m * Dn + nb] = xb;
  }
}

// ---------------- layernorm per row ----------------------------------------
__global__ __launch_bounds__(256) void k_ln(
    const float* __restrict__ X, const float* __restrict__ gamma,
    const float* __restrict__ beta, float* __restrict__ out) {
  int r = blockIdx.x;
  const float* x = X + (size_t)r * Dn;
  __shared__ float rs[256], rq[256];
  float vals[4], s = 0.f, ss = 0.f;
#pragma unroll
  for (int j = 0; j < 4; ++j) {
    float vv = x[threadIdx.x + 256 * j];
    vals[j] = vv; s += vv; ss += vv * vv;
  }
  rs[threadIdx.x] = s; rq[threadIdx.x] = ss;
  __syncthreads();
  for (int o = 128; o > 0; o >>= 1) {
    if (threadIdx.x < o) { rs[threadIdx.x] += rs[threadIdx.x + o]; rq[threadIdx.x] += rq[threadIdx.x + o]; }
    __syncthreads();
  }
  float mu = rs[0] * (1.f / Dn);
  float var = rq[0] * (1.f / Dn) - mu * mu;
  float inv = rsqrtf(var + 1e-6f);
#pragma unroll
  for (int j = 0; j < 4; ++j) {
    int n = threadIdx.x + 256 * j;
    out[(size_t)r * Dn + n] = (vals[j] - mu) * inv * gamma[n] + beta[n];
  }
}

extern "C" void kernel_launch(void* const* d_in, const int* in_sizes, int n_in,
                              void* d_out, int out_size, void* d_ws, size_t ws_size,
                              hipStream_t stream) {
  const float* q     = (const float*)d_in[0];
  const float* k     = (const float*)d_in[1];
  const float* v     = (const float*)d_in[2];
  const float* Wq    = (const float*)d_in[3];
  const float* Wk    = (const float*)d_in[4];
  const float* Wv    = (const float*)d_in[5];
  const float* ws    = (const float*)d_in[6];
  const float* Wo    = (const float*)d_in[7];
  const float* bo    = (const float*)d_in[8];
  const float* gamma = (const float*)d_in[9];
  const float* beta  = (const float*)d_in[10];
  const unsigned char* mask = (const unsigned char*)d_in[11];

  float* out  = (float*)d_out;                         // (B,L,D)
  float* attn = out + (size_t)Bn * Ln * Dn;            // (B,H,L,L)

  float* wsf = (float*)d_ws;
  const size_t PROJ = (size_t)Bn * Hn * Ln * 64;       // 4194304
  float* qh  = wsf;
  float* kh  = qh  + PROJ;
  float* vh  = kh  + PROJ;
  float* ctx = vh  + PROJ;
  float* tmp = ctx + PROJ;
  float* sq  = tmp + PROJ;
  float* sk  = sq  + (size_t)Bn * Hn * Ln;

  k_proj <<<dim3(16, 64, 3), 256, 0, stream>>>(q, k, v, Wq, Wk, Wv, qh, kh, vh);
  k_logit<<<512, 256, 0, stream>>>(qh, kh, ws, sq, sk);
  k_attn <<<dim3(Ln / 32, Bn * Hn), 256, 0, stream>>>(sq, sk, vh, mask, attn, ctx);
  k_ogemm<<<dim3(16, 64), 256, 0, stream>>>(ctx, Wo, bo, q, tmp);
  k_ln   <<<Bn * Ln, 256, 0, stream>>>(tmp, gamma, beta, out);
}
